// Attention_43782896615765
// MI455X (gfx1250) — matrix operand order repaired and checked by reading, not compiled, not run on hardware
//
#include <hip/hip_runtime.h>

// Flash attention for MI455X (gfx1250, wave32, WMMA bf16 16x16x32, f32 accum)
// with Tensor-Data-Mover staging of pre-converted bf16 K^T / V tiles.
// B=16, QT=KT=2048, D=1024. Br=32 q-rows/block, Bc=64 k-rows/iteration.

typedef __attribute__((ext_vector_type(16))) __bf16 v16bf;
typedef __attribute__((ext_vector_type(8)))  float  v8f;
typedef __attribute__((ext_vector_type(4)))  unsigned int v4u;
typedef __attribute__((ext_vector_type(8)))  int    v8i;
typedef __attribute__((ext_vector_type(4)))  int    v4i;

#define NB    16
#define SQT   2048
#define SKT   2048
#define SD    1024
#define BR    32
#define BC    64
#define NKB   (SKT / BC)      // 32 k-blocks
#define KD_STEPS (SD / 32)    // 32 wmma k-steps for the score GEMM
#define KT_STRIDE 72          // bf16 stride of K^T tile rows (64 data + 8 pad) = 144B
#define V_STRIDE  136         // bf16 stride of V sub-tile rows (128 data + 8 pad) = 272B
#define P_STRIDE  72
#define L2E   1.44269504088896340736f
#define MASKV (-4294967295.0f)   // -2^32+1, as in the reference

// LDS carve (bytes)
#define OFF_KT  0
#define OFF_V   (SD * KT_STRIDE * 2)                   // 147456
#define OFF_P   (OFF_V + 8 * BC * V_STRIDE * 2)        // + 139264 = 286720
#define OFF_ST  (OFF_P + BR * P_STRIDE * 2)            // + 4608   = 291328
#define LDS_BYTES (OFF_ST + (128 + 128 + 32 + 32 + 32) * 4)  // 292736

// workspace layout for bf16 prepass copies
#define KT16_BYTES ((size_t)NB * SD * SKT * 2)   // K transposed [b][d][k]  (64 MiB)
#define V16_BYTES  ((size_t)NB * SKT * SD * 2)   // V row-major  [b][k][d]  (64 MiB)
#define WS_NEED    (KT16_BYTES + V16_BYTES)

#define SWZ_XOR(x, m) \
  __int_as_float(__builtin_amdgcn_ds_swizzle(__float_as_int(x), ((m) << 10) | 0x1f))

#if defined(__has_builtin)
#if __has_builtin(__builtin_amdgcn_tensor_load_to_lds)
#define HAVE_TDM 1
#endif
#endif
#ifndef HAVE_TDM
#define HAVE_TDM 0
#endif

union Frag {
  v16bf bf;
  unsigned u[8];
  uint4 q[2];
};

__device__ __forceinline__ unsigned bf16u(float f) {
  unsigned x = __float_as_uint(f);
  return (x + 0x7FFFu + ((x >> 16) & 1u)) >> 16;   // RNE f32->bf16
}
__device__ __forceinline__ unsigned pkbf(float lo, float hi) {
  return bf16u(lo) | (bf16u(hi) << 16);
}

#if HAVE_TDM
// Issue one 2-D TDM tile load. g0: {flags, lds_addr, gaddr_lo, gaddr_hi|type};
// g1 packs dims/strides/tile + LDS pad fields per the CDNA5 D# spec.
__device__ __forceinline__ void tdm_load_2d(unsigned lds_off, const void* gaddr,
                                            unsigned g1w0, unsigned dim0, unsigned dim1,
                                            unsigned tile0, unsigned tile1,
                                            unsigned stride0) {
  unsigned long long ga = (unsigned long long)gaddr;
  v4u g0 = { 1u,                                     // count=1, user descriptor
             lds_off,                                // lds_addr (bytes)
             (unsigned)(ga & 0xffffffffu),
             (unsigned)((ga >> 32) & 0x01ffffffu) | 0x80000000u };  // type=2 ("image")
  v8i g1 = { (int)g1w0,                              // data_size/pad fields
             (int)(dim0 << 16),                      // tensor_dim0[15:0]
             (int)((dim0 >> 16) | (dim1 << 16)),     // dim0 hi | tensor_dim1[15:0]
             (int)((dim1 >> 16) | (tile0 << 16)),    // dim1 hi | tile_dim0
             (int)tile1,                             // tile_dim1 (tile_dim2=0)
             (int)stride0,                           // tensor_dim0_stride lo
             0, 0 };
  v4i z4 = { 0, 0, 0, 0 };
#if __clang_major__ >= 23
  v8i z8 = { 0, 0, 0, 0, 0, 0, 0, 0 };
  __builtin_amdgcn_tensor_load_to_lds(g0, g1, z4, z4, z8, 0);
#else
  __builtin_amdgcn_tensor_load_to_lds(g0, g1, z4, z4, 0);
#endif
}
#endif

// g1 word0: data_size=2B (code 1), pad_enable, pad_interval code, pad_amount code
#define G1W0(ivl) ((1u << 16) | (1u << 20) | ((unsigned)(ivl) << 22) | (3u << 25))

template <bool USE_TDM>
__global__ __launch_bounds__(256, 1) void fattn_cdna5(
    const float* __restrict__ Q, const float* __restrict__ K,
    const float* __restrict__ V, const int* __restrict__ QL,
    const int* __restrict__ KL, float* __restrict__ Out,
    const unsigned short* __restrict__ Kt16, const unsigned short* __restrict__ V16) {
  extern __shared__ char smem[];
  unsigned short* ktS = (unsigned short*)(smem + OFF_KT); // K^T tile [SD][KT_STRIDE]
  unsigned short* vS  = (unsigned short*)(smem + OFF_V);  // 8 sub-tiles [BC][V_STRIDE]
  unsigned short* pS  = (unsigned short*)(smem + OFF_P);  // P tile [BR][P_STRIDE]
  float* wmaxS = (float*)(smem + OFF_ST);
  float* wsumS = wmaxS + 128;
  float* mS    = wsumS + 128;
  float* lS    = mS + 32;
  float* cS    = lS + 32;

  const int tid  = threadIdx.x;
  const int lane = tid & 31;
  const int wave = tid >> 5;
  const int rw   = wave & 1;
  const int cw   = wave >> 1;
  const int hi8  = (lane & 16) >> 1;
  const int l15  = lane & 15;

  const int b    = blockIdx.y;
  const int q0   = blockIdx.x * BR;
  const int qlen = QL[b];
  const int klen = KL[b];

  const float* Qb = Q + (size_t)b * SQT * SD;
  const float* Kb = K + (size_t)b * SKT * SD;
  const float* Vb = V + (size_t)b * SKT * SD;
  const unsigned short* KtB = Kt16 + (size_t)b * SD * SKT;  // [d][k], pitch SKT
  const unsigned short* VbB = V16 + (size_t)b * SKT * SD;   // [k][d], pitch SD

  // ---- Q tile as resident WMMA A-fragments ----
  Frag qf[KD_STEPS];
  {
    const int qr = q0 + rw * 16 + l15;
    const float* qrow = Qb + (size_t)qr * SD;
#pragma unroll
    for (int f = 0; f < KD_STEPS; ++f) {
      const int kd0 = f * 32 + ((lane & 16) ? 8 : 0);
      float4 a0 = *(const float4*)(qrow + kd0);
      float4 a1 = *(const float4*)(qrow + kd0 + 4);
      float4 b0 = *(const float4*)(qrow + kd0 + 16);
      float4 b1 = *(const float4*)(qrow + kd0 + 20);
      qf[f].u[0] = pkbf(a0.x, a0.y); qf[f].u[1] = pkbf(a0.z, a0.w);
      qf[f].u[2] = pkbf(a1.x, a1.y); qf[f].u[3] = pkbf(a1.z, a1.w);
      qf[f].u[4] = pkbf(b0.x, b0.y); qf[f].u[5] = pkbf(b0.z, b0.w);
      qf[f].u[6] = pkbf(b1.x, b1.y); qf[f].u[7] = pkbf(b1.z, b1.w);
    }
  }

  if (tid < 32) { mS[tid] = -__builtin_inff(); lS[tid] = 0.0f; }

  v8f zero8 = {0.f, 0.f, 0.f, 0.f, 0.f, 0.f, 0.f, 0.f};
  v8f o[2][8];
#pragma unroll
  for (int mt = 0; mt < 2; ++mt)
#pragma unroll
    for (int nt = 0; nt < 8; ++nt) o[mt][nt] = zero8;

  for (int j = 0; j < NKB; ++j) {
    __syncthreads();  // previous iteration done reading LDS

    if (USE_TDM) {
#if HAVE_TDM
      if (wave == 0) {
        // K^T tile: 64 (k, contiguous) x 1024 (d rows); 128B rows + 16B LDS pad
        tdm_load_2d(OFF_KT, KtB + (size_t)j * BC, G1W0(4),
                    SKT, SD, BC, SD, SKT);
        // V: eight 128(d) x 64(k) sub-tiles; 256B rows + 16B LDS pad
#pragma unroll
        for (int c = 0; c < 8; ++c)
          tdm_load_2d(OFF_V + c * (BC * V_STRIDE * 2),
                      VbB + (size_t)j * BC * SD + c * 128, G1W0(5),
                      SD, SKT, 128, BC, SD);
#if __has_builtin(__builtin_amdgcn_s_wait_tensorcnt)
        __builtin_amdgcn_s_wait_tensorcnt(0);
#else
        asm volatile("s_wait_tensorcnt 0x0" ::: "memory");
#endif
      }
#endif
      __syncthreads();
      // prefetch next block's bf16 tiles into L2 while we compute
      if (j + 1 < NKB) {
        const char* nk = (const char*)(KtB + (size_t)(j + 1) * BC);
        const char* nv = (const char*)(VbB + (size_t)(j + 1) * BC * SD);
#pragma unroll
        for (int i = 0; i < 4; ++i) {
          __builtin_prefetch(nk + (size_t)(tid + i * 256) * (SKT * 2), 0, 1);
          __builtin_prefetch(nv + (size_t)(tid + i * 256) * 128, 0, 1);
        }
      }
    } else {
      // ---- fallback: stage f32 K/V from global, convert to bf16 in LDS ----
      const float* kt0 = Kb + (size_t)j * BC * SD;
      const float* vt0 = Vb + (size_t)j * BC * SD;
#pragma unroll 4
      for (int it = 0; it < (BC * SD) / (256 * 4); ++it) {
        int flat = (it * 256 + tid) * 4;
        int k = flat >> 10;
        int d = flat & (SD - 1);
        float4 kv = *(const float4*)(kt0 + (size_t)k * SD + d);
        ktS[(d + 0) * KT_STRIDE + k] = (unsigned short)bf16u(kv.x);
        ktS[(d + 1) * KT_STRIDE + k] = (unsigned short)bf16u(kv.y);
        ktS[(d + 2) * KT_STRIDE + k] = (unsigned short)bf16u(kv.z);
        ktS[(d + 3) * KT_STRIDE + k] = (unsigned short)bf16u(kv.w);
        float4 vv = *(const float4*)(vt0 + (size_t)k * SD + d);
        int c = d >> 7, ln = d & 127;
        *(uint2*)(vS + (c * BC + k) * V_STRIDE + ln) =
            make_uint2(pkbf(vv.x, vv.y), pkbf(vv.z, vv.w));
      }
      __syncthreads();
      if (j + 1 < NKB) {
        const char* nk = (const char*)(Kb + (size_t)(j + 1) * BC * SD);
        const char* nv = (const char*)(Vb + (size_t)(j + 1) * BC * SD);
#pragma unroll
        for (int i = 0; i < 8; ++i) {
          __builtin_prefetch(nk + (size_t)(i * 256 + tid) * 128, 0, 1);
          __builtin_prefetch(nv + (size_t)(i * 256 + tid) * 128, 0, 1);
        }
      }
    }

    // ---- scores: this wave's 16x16 tile of S = Q_blk . K_blk^T ----
    v8f s = zero8;
    {
      const int n0 = cw * 16;
#pragma unroll
      for (int f = 0; f < KD_STEPS; ++f) {
        Frag kf;  // B-fragment: lane<->N column, dwords<->K pairs (table order)
        const unsigned short* src = ktS + (f * 32 + hi8) * KT_STRIDE + n0 + l15;
#pragma unroll
        for (int i = 0; i < 8; ++i) {
          const int e = (i < 4) ? 2 * i : 2 * i + 8;
          kf.u[i] = (unsigned)src[e * KT_STRIDE] | ((unsigned)src[(e + 1) * KT_STRIDE] << 16);
        }
        s = __builtin_amdgcn_wmma_f32_16x16x32_bf16(false, qf[f].bf, false, kf.bf,
                                                    (short)0, s, false, false);
      }
    }

    // ---- mask + in-tile row max ----
    const int kg = j * BC + cw * 16 + l15;
#pragma unroll
    for (int v = 0; v < 8; ++v) {
      int qg = q0 + rw * 16 + v + hi8;
      bool valid = (qg < qlen) && (kg < klen);
      s[v] = valid ? s[v] : MASKV;
      float t = s[v];
      t = fmaxf(t, SWZ_XOR(t, 1));
      t = fmaxf(t, SWZ_XOR(t, 2));
      t = fmaxf(t, SWZ_XOR(t, 4));
      t = fmaxf(t, SWZ_XOR(t, 8));
      if (lane == 0)       wmaxS[cw * 32 + rw * 16 + v]     = t;
      else if (lane == 16) wmaxS[cw * 32 + rw * 16 + 8 + v] = t;
    }
    __syncthreads();

    // ---- P = exp(S - m_new), in-tile row sums, P -> LDS (bf16) ----
#pragma unroll
    for (int v = 0; v < 8; ++v) {
      int row = rw * 16 + v + hi8;
      float mn = mS[row];
      mn = fmaxf(mn, wmaxS[row]);
      mn = fmaxf(mn, wmaxS[32 + row]);
      mn = fmaxf(mn, wmaxS[64 + row]);
      mn = fmaxf(mn, wmaxS[96 + row]);
      float p = exp2f((s[v] - mn) * L2E);
      s[v] = p;
      float t = p;
      t += SWZ_XOR(t, 1);
      t += SWZ_XOR(t, 2);
      t += SWZ_XOR(t, 4);
      t += SWZ_XOR(t, 8);
      if (lane == 0)       wsumS[cw * 32 + rw * 16 + v]     = t;
      else if (lane == 16) wsumS[cw * 32 + rw * 16 + 8 + v] = t;
      pS[row * P_STRIDE + cw * 16 + l15] = (unsigned short)bf16u(p);
    }
    __syncthreads();

    // ---- update running stats m, l and rescale factor c ----
    if (tid < 32) {
      float mo = mS[tid];
      float mn = fmaxf(fmaxf(wmaxS[tid], wmaxS[32 + tid]),
                       fmaxf(wmaxS[64 + tid], wmaxS[96 + tid]));
      mn = fmaxf(mn, mo);
      float c = exp2f((mo - mn) * L2E);
      lS[tid] = lS[tid] * c +
                (wsumS[tid] + wsumS[32 + tid] + wsumS[64 + tid] + wsumS[96 + tid]);
      mS[tid] = mn;
      cS[tid] = c;
    }
    __syncthreads();

    // ---- rescale O, then O += P . V on this wave's 32x128 slab ----
#pragma unroll
    for (int mt = 0; mt < 2; ++mt)
#pragma unroll
      for (int v = 0; v < 8; ++v) {
        float c = cS[mt * 16 + v + hi8];
#pragma unroll
        for (int nt = 0; nt < 8; ++nt) o[mt][nt][v] *= c;
      }

    Frag pf[2][2];
#pragma unroll
    for (int mt = 0; mt < 2; ++mt)
#pragma unroll
      for (int kk = 0; kk < 2; ++kk) {
        const int row = mt * 16 + l15;
        const int kd0 = kk * 32 + ((lane & 16) ? 8 : 0);
        const unsigned short* src = pS + row * P_STRIDE + kd0;
        pf[mt][kk].q[0] = *(const uint4*)(src);
        pf[mt][kk].q[1] = *(const uint4*)(src + 16);
      }

#pragma unroll
    for (int kk = 0; kk < 2; ++kk) {
#pragma unroll
      for (int nt = 0; nt < 8; ++nt) {
        Frag vf;  // B-fragment from this wave's V sub-tile
        const unsigned short* src =
            vS + (wave * BC + kk * 32 + hi8) * V_STRIDE + nt * 16 + l15;
#pragma unroll
        for (int i = 0; i < 8; ++i) {
          const int e = (i < 4) ? 2 * i : 2 * i + 8;
          vf.u[i] = (unsigned)src[e * V_STRIDE] | ((unsigned)src[(e + 1) * V_STRIDE] << 16);
        }
        o[0][nt] = __builtin_amdgcn_wmma_f32_16x16x32_bf16(false, pf[0][kk].bf, false, vf.bf,
                                                           (short)0, o[0][nt], false, false);
        o[1][nt] = __builtin_amdgcn_wmma_f32_16x16x32_bf16(false, pf[1][kk].bf, false, vf.bf,
                                                           (short)0, o[1][nt], false, false);
      }
    }
  }

  // ---- epilogue: O /= l, write out ----
  float* Ob = Out + (size_t)b * SQT * SD + (size_t)q0 * SD;
#pragma unroll
  for (int mt = 0; mt < 2; ++mt)
#pragma unroll
    for (int v = 0; v < 8; ++v) {
      int row = mt * 16 + v + hi8;
      float inv = 1.0f / lS[row];
      float* orow = Ob + (size_t)row * SD + wave * 128 + l15;
#pragma unroll
      for (int nt = 0; nt < 8; ++nt) orow[nt * 16] = o[mt][nt][v] * inv;
    }
}

// ---- prepass: K (f32, [b][k][d]) -> bf16 transposed [b][d][k], LDS-tiled ----
__global__ __launch_bounds__(256) void prep_kT(const float* __restrict__ K,
                                               unsigned short* __restrict__ Kt16) {
  __shared__ unsigned short t[64 * 65];
  const int b  = blockIdx.z;
  const int k0 = blockIdx.x * 64;
  const int d0 = blockIdx.y * 64;
  const float* src = K + (size_t)b * SKT * SD;
#pragma unroll
  for (int it = 0; it < 16; ++it) {
    int idx = it * 256 + threadIdx.x;
    int k = idx >> 6, d = idx & 63;
    t[d * 65 + k] = (unsigned short)bf16u(src[(size_t)(k0 + k) * SD + d0 + d]);
  }
  __syncthreads();
  unsigned short* dst = Kt16 + (size_t)b * SD * SKT;
#pragma unroll
  for (int it = 0; it < 8; ++it) {
    int idx2 = (it * 256 + threadIdx.x) * 2;
    int d = idx2 >> 6, k = idx2 & 63;
    unsigned w = (unsigned)t[d * 65 + k] | ((unsigned)t[d * 65 + k + 1] << 16);
    *(unsigned*)(dst + (size_t)(d0 + d) * SKT + k0 + k) = w;
  }
}

// ---- prepass: V (f32) -> bf16, streaming ----
__global__ __launch_bounds__(256) void prep_v(const float* __restrict__ V,
                                              unsigned short* __restrict__ V16) {
  size_t i = ((size_t)blockIdx.x * 256 + threadIdx.x) * 4;
  float4 v = *(const float4*)(V + i);
  *(uint2*)(V16 + i) = make_uint2(pkbf(v.x, v.y), pkbf(v.z, v.w));
}

extern "C" void kernel_launch(void* const* d_in, const int* in_sizes, int n_in,
                              void* d_out, int out_size, void* d_ws, size_t ws_size,
                              hipStream_t stream) {
  const float* Q = (const float*)d_in[0];
  const float* K = (const float*)d_in[1];
  const float* V = (const float*)d_in[2];
  const int* QL  = (const int*)d_in[3];
  const int* KL  = (const int*)d_in[4];
  float* O = (float*)d_out;

  dim3 grid(SQT / BR, NB);  // q-block fastest -> same-batch blocks adjacent (L2 reuse)
  dim3 block(256);

  if (HAVE_TDM && ws_size >= WS_NEED) {
    unsigned short* Kt16 = (unsigned short*)d_ws;
    unsigned short* V16  = (unsigned short*)((char*)d_ws + KT16_BYTES);
    prep_kT<<<dim3(SKT / 64, SD / 64, NB), block, 0, stream>>>(K, Kt16);
    prep_v<<<dim3((unsigned)((size_t)NB * SKT * SD / (256 * 4))), block, 0, stream>>>(V, V16);
    fattn_cdna5<true><<<grid, block, LDS_BYTES, stream>>>(Q, K, V, QL, KL, O, Kt16, V16);
  } else {
    fattn_cdna5<false><<<grid, block, LDS_BYTES, stream>>>(Q, K, V, QL, KL, O,
                                                           (const unsigned short*)nullptr,
                                                           (const unsigned short*)nullptr);
  }
  (void)in_sizes; (void)n_in; (void)out_size;
}
